// AntiSymmetric_14130442404420
// MI455X (gfx1250) — matrix operand order repaired and checked by reading, not compiled, not run on hardware
//
#include <hip/hip_runtime.h>
#include <hip/hip_bf16.h>
#include <math.h>

// ---------------------------------------------------------------------------
// AntiSymmetricConv x2 + leaky_relu + softmax for MI455X (gfx1250, wave32)
// GEMMs use v_wmma_f32_16x16x32_bf16; edge aggregation uses float4 gathers +
// global f32 atomics (L2-resident working set: ~100MB << 192MB L2).
// ---------------------------------------------------------------------------

#define F_DIM 128
#define EPSILON 0.1f
#define GAMMA 0.1f

typedef __bf16 bf16_t;
typedef __attribute__((ext_vector_type(16))) __bf16 bf16x16;
typedef __attribute__((ext_vector_type(8)))  __bf16 bf16x8;
typedef __attribute__((ext_vector_type(8)))  float  f32x8;

// ---------------- small utility kernels ----------------

__global__ void k_fill_f32(float* __restrict__ p, float v, long n) {
    long i = (long)blockIdx.x * blockDim.x + threadIdx.x;
    long stride = (long)gridDim.x * blockDim.x;
    for (; i < n; i += stride) p[i] = v;
}

__global__ void k_count_deg(const long long* __restrict__ dst, float* __restrict__ deg, int E) {
    int e = blockIdx.x * blockDim.x + threadIdx.x;
    if (e < E) atomicAdd(&deg[(int)dst[e]], 1.0f);
}

__global__ void k_rsqrt_inplace(float* __restrict__ d, int n) {
    int i = blockIdx.x * blockDim.x + threadIdx.x;
    if (i < n) d[i] = rsqrtf(d[i]);
}

__global__ void k_f32_to_bf16(const float* __restrict__ src, bf16_t* __restrict__ dst, long n) {
    long i = (long)blockIdx.x * blockDim.x + threadIdx.x;
    long stride = (long)gridDim.x * blockDim.x;
    for (; i < n; i += stride) dst[i] = (bf16_t)src[i];
}

// aw = W - W^T - gamma*I  (128x128), converted to bf16 row-major
__global__ void k_make_aw_bf16(const float* __restrict__ W, bf16_t* __restrict__ awbf) {
    int t = blockIdx.x * blockDim.x + threadIdx.x;   // 0 .. 128*128-1
    int i = t >> 7, j = t & (F_DIM - 1);
    float v = W[i * F_DIM + j] - W[j * F_DIM + i] - ((i == j) ? GAMMA : 0.0f);
    awbf[t] = (bf16_t)v;
}

// ---------------- WMMA GEMM:  C[N x 128] = A[N x 128] * B[128 x 128]^T ------
// A, B row-major bf16.  C[m][j] = sum_k A[m][k] * B[j][k].
// One wave per 16x16 output tile; block = 256 threads = 8 waves = all 8 jTiles
// of one 16-row strip.  K loop: 4 x v_wmma_f32_16x16x32_bf16.

__global__ void __launch_bounds__(256)
k_gemm_bf16_nt(const bf16_t* __restrict__ A, const bf16_t* __restrict__ B,
               float* __restrict__ C, int N) {
    const int lane = threadIdx.x & 31;
    const int wv   = threadIdx.x >> 5;          // jTile = wv (8 tiles of 16 cols)
    const int l16  = lane & 15;
    const int hi   = lane >> 4;                 // 0: lanes 0-15, 1: lanes 16-31

    const int mRow  = blockIdx.x * 16 + l16;    // A row this lane loads
    const int kbA   = hi * 16;                  // A: K base 0 or 16
    const int bRow  = wv * 16 + l16;            // W row (= output column)
    const int kbB   = hi * 16;                  // B: K base 0 or 16

    const bf16_t* Arow = A + (size_t)mRow * F_DIM;
    const bf16_t* Brow = B + (size_t)bRow * F_DIM;

    f32x8 c = {};
    #pragma unroll
    for (int kt = 0; kt < F_DIM; kt += 32) {
        bf16x8 a0 = *(const bf16x8*)(Arow + kt + kbA);        // K = kb .. kb+7
        bf16x8 a1 = *(const bf16x8*)(Arow + kt + kbA + 8);    // K = kb+8 .. kb+15
        bf16x16 a = __builtin_shufflevector(a0, a1,
                        0,1,2,3,4,5,6,7,8,9,10,11,12,13,14,15);
        bf16x16 b = *(const bf16x16*)(Brow + kt + kbB);       // 16 contiguous K
        c = __builtin_amdgcn_wmma_f32_16x16x32_bf16(
                false, a, false, b, (short)0, c, false, false);
    }

    // C/D layout: VGPR r -> row M = r (+8 for lanes>=16), col N = lane%16
    const int row0 = blockIdx.x * 16 + hi * 8;
    const int col  = wv * 16 + l16;
    float* Cb = C + (size_t)row0 * F_DIM + col;
    #pragma unroll
    for (int r = 0; r < 8; ++r) Cb[(size_t)r * F_DIM] = c[r];
}

// ---------------- edge aggregation ----------------
// agg[dst] += h[src] * dinv[src]*dinv[dst]   (one wave per edge, float4/lane)

__global__ void k_edge_agg(const long long* __restrict__ src,
                           const long long* __restrict__ dst,
                           const float* __restrict__ h,
                           const float* __restrict__ dinv,
                           float* __restrict__ agg, int E) {
    long t = (long)blockIdx.x * blockDim.x + threadIdx.x;
    int e = (int)(t >> 5);
    if (e >= E) return;
    int f4 = (int)(t & 31) * 4;
    int s = (int)src[e];
    int d = (int)dst[e];
    float norm = dinv[s] * dinv[d];
    const float4 v = *reinterpret_cast<const float4*>(h + (size_t)s * F_DIM + f4);
    float* o = agg + (size_t)d * F_DIM + f4;
    atomicAdd(o + 0, v.x * norm);
    atomicAdd(o + 1, v.y * norm);
    atomicAdd(o + 2, v.z * norm);
    atomicAdd(o + 3, v.w * norm);
}

// ---------------- layer epilogue ----------------
// r = xin + eps*tanh(z + agg + h*dinv^2 + bias)
// layer1: out = xin + leaky_relu(r)   (input to layer 2)
// layer2: out = r                     (softmax input)

__global__ void k_epilogue(const float* __restrict__ xin, const float* __restrict__ h,
                           const float* __restrict__ z,   const float* __restrict__ agg,
                           const float* __restrict__ dinv,const float* __restrict__ bias,
                           float* __restrict__ out, long NF, int layer1) {
    long i = (long)blockIdx.x * blockDim.x + threadIdx.x;
    if (i >= NF) return;
    int n = (int)(i >> 7);
    int f = (int)(i & (F_DIM - 1));
    float di = dinv[n];
    float v = z[i] + agg[i] + h[i] * di * di + bias[f];
    float r = xin[i] + EPSILON * tanhf(v);
    if (layer1) {
        float y = (r > 0.0f) ? r : 0.01f * r;   // leaky_relu(0.01)
        out[i] = xin[i] + y;
    } else {
        out[i] = r;
    }
}

// ---------------- softmax (one wave32 per row, 4 vals/lane) ----------------

__global__ void k_softmax(const float* __restrict__ in, float* __restrict__ out, int N) {
    long t = (long)blockIdx.x * blockDim.x + threadIdx.x;
    int row = (int)(t >> 5);
    int lane = threadIdx.x & 31;
    if (row >= N) return;
    const float4 v = reinterpret_cast<const float4*>(in + (size_t)row * F_DIM)[lane];
    float m = fmaxf(fmaxf(v.x, v.y), fmaxf(v.z, v.w));
    #pragma unroll
    for (int o = 16; o > 0; o >>= 1) m = fmaxf(m, __shfl_xor(m, o, 32));
    float4 e;
    e.x = __expf(v.x - m); e.y = __expf(v.y - m);
    e.z = __expf(v.z - m); e.w = __expf(v.w - m);
    float s = e.x + e.y + e.z + e.w;
    #pragma unroll
    for (int o = 16; o > 0; o >>= 1) s += __shfl_xor(s, o, 32);
    float inv = 1.0f / s;
    float4 r; r.x = e.x * inv; r.y = e.y * inv; r.z = e.z * inv; r.w = e.w * inv;
    reinterpret_cast<float4*>(out + (size_t)row * F_DIM)[lane] = r;
}

// ---------------------------------------------------------------------------

extern "C" void kernel_launch(void* const* d_in, const int* in_sizes, int n_in,
                              void* d_out, int out_size, void* d_ws, size_t ws_size,
                              hipStream_t stream) {
    const float*      x    = (const float*)d_in[0];
    const float*      W1   = (const float*)d_in[1];
    const float*      b1   = (const float*)d_in[2];
    const float*      lin1 = (const float*)d_in[3];
    const float*      W2   = (const float*)d_in[4];
    const float*      b2   = (const float*)d_in[5];
    const float*      lin2 = (const float*)d_in[6];
    const long long*  ei   = (const long long*)d_in[7];

    const int N  = in_sizes[0] / F_DIM;
    const int E  = in_sizes[7] / 2;
    const long NF = (long)N * F_DIM;
    const long long* esrc = ei;
    const long long* edst = ei + E;

    // ---- workspace layout ----
    char* ws = (char*)d_ws;
    size_t off = 0;
    auto take = [&](size_t bytes) { void* p = ws + off; off += (bytes + 255) & ~255ull; return p; };
    float*  dinv = (float*)take((size_t)N * 4);
    bf16_t* xbf  = (bf16_t*)take((size_t)NF * 2);       // bf16 of current layer input
    bf16_t* wbf  = (bf16_t*)take((size_t)F_DIM * F_DIM * 2);
    float*  hbuf = (float*)take((size_t)NF * 4);        // GCN lin output / final pre-softmax
    float*  zbuf = (float*)take((size_t)NF * 4);        // antisymmetric term
    float*  agg  = (float*)take((size_t)NF * 4);        // edge aggregation
    float*  xin2 = (float*)take((size_t)NF * 4);        // layer-2 input

    const int T = 256;
    const int gNF    = (int)((NF + T - 1) / T);
    const int gE     = (E + T - 1) / T;
    const int gE32   = (int)(((long)E * 32 + T - 1) / T);
    const int gN     = (N + T - 1) / T;
    const int gGemm  = N / 16;                           // 50000/16 = 3125
    const int gFF    = (F_DIM * F_DIM) / T;
    const int gRow   = (int)(((long)N * 32 + T - 1) / T);

    // ---- degrees (shared by both layers): dinv = rsqrt(1 + indeg) ----
    k_fill_f32<<<gN, T, 0, stream>>>(dinv, 1.0f, N);
    k_count_deg<<<gE, T, 0, stream>>>(edst, dinv, E);
    k_rsqrt_inplace<<<gN, T, 0, stream>>>(dinv, N);

    // ================= layer 1 =================
    k_f32_to_bf16<<<gNF, T, 0, stream>>>(x, xbf, NF);

    // h = x @ lin1^T
    k_f32_to_bf16<<<gFF, T, 0, stream>>>(lin1, wbf, (long)F_DIM * F_DIM);
    k_gemm_bf16_nt<<<gGemm, T, 0, stream>>>(xbf, wbf, hbuf, N);

    // agg = scatter_sum(h[src]*norm -> dst)
    k_fill_f32<<<gNF, T, 0, stream>>>(agg, 0.0f, NF);
    k_edge_agg<<<gE32, T, 0, stream>>>(esrc, edst, hbuf, dinv, agg, E);

    // z = x @ (W1 - W1^T - gamma I)^T
    k_make_aw_bf16<<<gFF, T, 0, stream>>>(W1, wbf);
    k_gemm_bf16_nt<<<gGemm, T, 0, stream>>>(xbf, wbf, zbuf, N);

    // xin2 = x + leaky_relu(x + eps*tanh(z + agg + h*dinv^2 + b1))
    k_epilogue<<<gNF, T, 0, stream>>>(x, hbuf, zbuf, agg, dinv, b1, xin2, NF, 1);

    // ================= layer 2 =================
    k_f32_to_bf16<<<gNF, T, 0, stream>>>(xin2, xbf, NF);

    k_f32_to_bf16<<<gFF, T, 0, stream>>>(lin2, wbf, (long)F_DIM * F_DIM);
    k_gemm_bf16_nt<<<gGemm, T, 0, stream>>>(xbf, wbf, hbuf, N);

    k_fill_f32<<<gNF, T, 0, stream>>>(agg, 0.0f, NF);
    k_edge_agg<<<gE32, T, 0, stream>>>(esrc, edst, hbuf, dinv, agg, E);

    k_make_aw_bf16<<<gFF, T, 0, stream>>>(W2, wbf);
    k_gemm_bf16_nt<<<gGemm, T, 0, stream>>>(xbf, wbf, zbuf, N);

    // pre-softmax result into hbuf (reuse), then softmax -> d_out
    k_epilogue<<<gNF, T, 0, stream>>>(xin2, hbuf, zbuf, agg, dinv, b2, hbuf, NF, 0);
    k_softmax<<<gRow, T, 0, stream>>>(hbuf, (float*)d_out, N);
}
